// STGCN_25537875542346
// MI455X (gfx1250) — matrix operand order repaired and checked by reading, not compiled, not run on hardware
//
#include <hip/hip_runtime.h>
#include <hip/hip_bf16.h>

typedef __attribute__((ext_vector_type(16))) _Float16 v16h;
typedef __attribute__((ext_vector_type(8)))  _Float16 v8h;
typedef __attribute__((ext_vector_type(8)))  float    v8f;
typedef __attribute__((ext_vector_type(4)))  unsigned u32x4;
typedef __attribute__((ext_vector_type(8)))  unsigned u32x8;

#define B_   32
#define N_   64
#define T_   128
#define FIN  9
#define GH   64
#define LH   128
#define NG   512      // 4*LH gates
#define NC   16
#define DIN  4096     // N_*GH
#define M_   4096     // B_*T_
#define CNT  262144   // B_*N_*T_ (batchnorm population per channel)

__device__ __forceinline__ int lane_id() { return threadIdx.x & 31; }

__device__ __forceinline__ v8f wmma16(v16h a, v16h b, v8f c) {
  // D = A(16x32 f16) * B(32x16 f16) + C(16x16 f32)
  return __builtin_amdgcn_wmma_f32_16x16x32_f16(false, a, false, b, (short)0, c,
                                                false, false);
}

// --- CDNA5 async copy: per-lane 16B global -> LDS, tracked by ASYNCcnt -------
__device__ __forceinline__ void async_load_b128(void* lds, const void* g) {
  // generic LDS pointer: addr[31:0] is the LDS byte offset (ISA 10.2 aperture map)
  unsigned off = (unsigned)(size_t)lds;
  asm volatile("global_load_async_to_lds_b128 %0, %1, off"
               :: "v"(off), "v"(g) : "memory");
}
__device__ __forceinline__ void wait_async() {
  asm volatile("s_wait_asynccnt 0x0" ::: "memory");
}

// --- CDNA5 Tensor Data Mover: 2D f16 tile global -> LDS (ISA ch. 8 D#) ------
// rows x cols elements, row stride = `stride` elements, 2 bytes each.
__device__ __forceinline__ void tdm_load_tile_f16(void* lds, const void* g,
                                                  unsigned rows, unsigned cols,
                                                  unsigned stride) {
  unsigned lds_off = __builtin_amdgcn_readfirstlane((unsigned)(size_t)lds);
  unsigned long long ga = (unsigned long long)(size_t)g;
  unsigned ga_lo = __builtin_amdgcn_readfirstlane((unsigned)ga);
  unsigned ga_hi = __builtin_amdgcn_readfirstlane((unsigned)(ga >> 32));
  u32x4 g0;
  g0[0] = 1u;                                    // count=1, user descriptor
  g0[1] = lds_off;                               // lds_addr  [63:32]
  g0[2] = ga_lo;                                 // global_addr [95:64]
  g0[3] = (ga_hi & 0x01FFFFFFu) | (2u << 30);    // addr[120:96] | type=2
  u32x8 g1;
  g1[0] = 1u << 16;                              // data_size code 1 = 2 bytes
  g1[1] = cols << 16;                            // tensor_dim0[15:0]  @ [79:48]
  g1[2] = rows << 16;                            // tensor_dim1[15:0]  @ [111:80]
  g1[3] = cols << 16;                            // tile_dim0 @ [127:112]
  g1[4] = rows;                                  // tile_dim1 @ [143:128], tile_dim2=0
  g1[5] = stride;                                // tensor_dim0_stride[31:0]
  g1[6] = 0u;                                    // stride hi, dim1_stride lo
  g1[7] = 0u;                                    // dim1_stride hi
  asm volatile("tensor_load_to_lds %0, %1" :: "s"(g0), "s"(g1) : "memory");
}

// A fragment (16x32 f16) from row-major [m][k] storage (ISA 7.12.2):
// lane<16: row=lane, K-base 0; lane>=16: row=lane-16, K-base 8.
// lane holds two contiguous 8-wide K runs: kb+0..7 and kb+16..23.
__device__ __forceinline__ v16h pack_a_rm(const _Float16* p, int ld, int row0, int k0) {
  int lane = lane_id();
  const _Float16* base = p + (row0 + (lane & 15)) * ld + k0 + ((lane & 16) ? 8 : 0);
  v8h lo = *(const v8h*)(base);
  v8h hi = *(const v8h*)(base + 16);
  return __builtin_shufflevector(lo, hi, 0, 1, 2, 3, 4, 5, 6, 7,
                                 8, 9, 10, 11, 12, 13, 14, 15);
}

// B fragment (32x16 f16) from (N,K) row-major storage (column-of-B = row here):
// lanes 0-15 K 0..7,16..23; lanes 16-31 K 8..15,24..31 (same order as A).
__device__ __forceinline__ v16h pack_b_nk(const _Float16* p, int ldk, int k0, int col0) {
  int lane = lane_id();
  const _Float16* base = p + (col0 + (lane & 15)) * ldk + k0 + ((lane & 16) ? 8 : 0);
  v8h lo = *(const v8h*)(base);
  v8h hi = *(const v8h*)(base + 16);
  return __builtin_shufflevector(lo, hi, 0, 1, 2, 3, 4, 5, 6, 7,
                                 8, 9, 10, 11, 12, 13, 14, 15);
}

// ---------------- adjacency normalization (64x64) ----------------------------
__global__ void adjnorm_kernel(const float* __restrict__ adj,
                               _Float16* __restrict__ adj16) {
  __shared__ float dinv[N_];
  int i = threadIdx.x;  // 64 threads
  float deg = 1.0f;     // self loop
  for (int j = 0; j < N_; ++j) deg += adj[i * N_ + j];
  dinv[i] = (deg > 0.f) ? rsqrtf(deg) : 0.f;
  __syncthreads();
  float di = dinv[i];
  for (int j = 0; j < N_; ++j) {
    float a = adj[i * N_ + j] + ((i == j) ? 1.f : 0.f);
    adj16[i * N_ + j] = (_Float16)(a * di * dinv[j]);
  }
}

__global__ void f32_to_f16_kernel(const float* __restrict__ src,
                                  _Float16* __restrict__ dst, int n) {
  int i = blockIdx.x * blockDim.x + threadIdx.x;
  if (i < n) dst[i] = (_Float16)src[i];
}

// ---------------- fused GCN layer: H = Anorm @ (X @ W), per (b,t) slice ------
// grid = B_*T_, block = 128 (4 waves); wave w owns a 16-row strip.
__global__ void gcn_kernel(const float* __restrict__ xsrc, int Fin,
                           const float* __restrict__ w,
                           const _Float16* __restrict__ adj16,
                           float* __restrict__ hout) {
  __shared__ __align__(32) _Float16 sX[N_ * 64];    // [row][k], zero padded
  __shared__ __align__(32) _Float16 sWt[GH * 64];   // [n][k]  (B operand layout)
  __shared__ __align__(32) _Float16 sAdj[N_ * N_];  // [m][n] row-major
  __shared__ __align__(32) _Float16 sXWt[GH * N_];  // [n][k]  (B operand layout)
  int bt = blockIdx.x;
  int bb = bt >> 7, t = bt & 127;
  int tid = threadIdx.x;
  int wv = tid >> 5;
  int lane = lane_id();
  // TDM: DMA the f16 adjacency tile (64x64, 8 KB) into LDS; wave 0 issues.
  if (tid < 32) {
    tdm_load_tile_f16(sAdj, adj16, N_, N_, N_);
    __builtin_amdgcn_s_wait_tensorcnt(0);
  }
  for (int idx = tid; idx < 4096; idx += 128) {
    int r = idx >> 6, c = idx & 63;
    float xv = (c < Fin) ? xsrc[(((size_t)bb * N_ + r) * T_ + t) * Fin + c] : 0.f;
    sX[idx] = (_Float16)xv;
    float wvv = (c < Fin) ? w[c * GH + r] : 0.f;  // sWt[n=r][k=c] = W[k][n]
    sWt[idx] = (_Float16)wvv;
  }
  __syncthreads();
  int ks1 = (Fin + 31) >> 5;  // 1 (Fin=9) or 2 (Fin=64)
  // XW = X * W  -> sXWt (written transposed, one 16B store per lane per tile)
#pragma unroll
  for (int ct = 0; ct < 4; ++ct) {
    v8f acc = {};
    for (int kc = 0; kc < ks1; ++kc) {
      v16h a = pack_a_rm(sX, 64, wv * 16, kc * 32);
      v16h b = pack_b_nk(sWt, 64, kc * 32, ct * 16);
      acc = wmma16(a, b, acc);
    }
    int col = ct * 16 + (lane & 15);
    int rb = wv * 16 + ((lane & 16) ? 8 : 0);
    v8h hv;
#pragma unroll
    for (int r = 0; r < 8; ++r) hv[r] = (_Float16)acc[r];
    *(v8h*)(sXWt + col * 64 + rb) = hv;   // [n=col][k=row]
  }
  __syncthreads();
  // H = Anorm * XW -> global (B,N,T,GH)
#pragma unroll
  for (int ct = 0; ct < 4; ++ct) {
    v8f acc = {};
#pragma unroll
    for (int kc = 0; kc < 2; ++kc) {
      v16h a = pack_a_rm(sAdj, N_, wv * 16, kc * 32);
      v16h b = pack_b_nk(sXWt, 64, kc * 32, ct * 16);
      acc = wmma16(a, b, acc);
    }
    int col = ct * 16 + (lane & 15);
    int rb = wv * 16 + ((lane & 16) ? 8 : 0);
#pragma unroll
    for (int r = 0; r < 8; ++r)
      hout[(((size_t)bb * N_ + rb + r) * T_ + t) * GH + col] = acc[r];
  }
}

// ---------------- deterministic BN stats: one block per channel --------------
__global__ void bn_stats_kernel(const float* __restrict__ h,
                                const float* __restrict__ gamma,
                                const float* __restrict__ beta,
                                float* __restrict__ cA, float* __restrict__ cB) {
  __shared__ float sS[256], sQ[256];
  int c = blockIdx.x;     // 0..63
  int tid = threadIdx.x;  // 256
  float s = 0.f, q = 0.f;
  for (int i = tid; i < CNT; i += 256) {
    float v = h[(size_t)i * 64 + c];
    s += v;
    q += v * v;
  }
  sS[tid] = s; sQ[tid] = q;
  __syncthreads();
  for (int st = 128; st > 0; st >>= 1) {
    if (tid < st) { sS[tid] += sS[tid + st]; sQ[tid] += sQ[tid + st]; }
    __syncthreads();
  }
  if (tid == 0) {
    float mean = sS[0] / (float)CNT;
    float var = sQ[0] / (float)CNT - mean * mean;
    float inv = rsqrtf(var + 1e-5f);
    float a = gamma[c] * inv;
    cA[c] = a;
    cB[c] = beta[c] - mean * a;
  }
}

__global__ void bn_relu_kernel(const float* __restrict__ h,
                               const float* __restrict__ cA,
                               const float* __restrict__ cB,
                               float* __restrict__ out) {
  size_t i = (size_t)blockIdx.x * blockDim.x + threadIdx.x;
  int c = (int)(i & 63);
  out[i] = fmaxf(h[i] * cA[c] + cB[c], 0.f);
}

// layer2 apply: hsum = relu(bn(h2)) + x1, packed f16 into (b,t, n*64+f) layout
__global__ void bn_relu_add_pack_kernel(const float* __restrict__ h2,
                                        const float* __restrict__ cA,
                                        const float* __restrict__ cB,
                                        const float* __restrict__ x1,
                                        _Float16* __restrict__ lstm_in) {
  size_t i = (size_t)blockIdx.x * blockDim.x + threadIdx.x;
  int f = (int)(i & 63);
  int t = (int)((i >> 6) & 127);
  int n = (int)((i >> 13) & 63);
  int b = (int)(i >> 19);
  float v = fmaxf(h2[i] * cA[f] + cB[f], 0.f) + x1[i];
  lstm_in[((size_t)b * T_ + t) * DIN + n * GH + f] = (_Float16)v;
}

// ---------------- LSTM input projection GEMM: C(M,512)=A(M,K)@W(512,K)^T + b -
// grid (M/64, 512/64), block 128 (4 waves); async-DMA tile staging into LDS.
__global__ void gemm_xp_kernel(const _Float16* __restrict__ A,
                               const _Float16* __restrict__ Bw,
                               const float* __restrict__ b1,
                               const float* __restrict__ b2,
                               float* __restrict__ C, int K) {
  __shared__ __align__(32) _Float16 sA[64 * 32];  // [m][k]
  __shared__ __align__(32) _Float16 sB[64 * 32];  // [n][k]
  int row0 = blockIdx.x * 64;
  int col0 = blockIdx.y * 64;
  int tid = threadIdx.x;
  int wv = tid >> 5;
  int lane = lane_id();
  v8f z = {};
  v8f acc[4] = {z, z, z, z};
  int r = tid >> 1;            // 0..63: tile row
  int half = (tid & 1) * 8;    // element offset 0 / 8 (covers k 0..7,16..23 / 8..15,24..31)
  const _Float16* gA = A + (size_t)(row0 + r) * K + half;
  const _Float16* gB = Bw + (size_t)(col0 + r) * K + half;
  _Float16* lA = sA + r * 32 + half;
  _Float16* lB = sB + r * 32 + half;
  for (int k0 = 0; k0 < K; k0 += 32) {
    async_load_b128(lA, gA + k0);
    async_load_b128(lA + 16, gA + k0 + 16);
    async_load_b128(lB, gB + k0);
    async_load_b128(lB + 16, gB + k0 + 16);
    if (k0 + 32 < K) {  // pull next chunk toward L2/L0
      __builtin_prefetch(gA + k0 + 32, 0, 1);
      __builtin_prefetch(gB + k0 + 32, 0, 1);
    }
    wait_async();
    __syncthreads();
    v16h a = pack_a_rm(sA, 32, wv * 16, 0);
#pragma unroll
    for (int ct = 0; ct < 4; ++ct)
      acc[ct] = wmma16(a, pack_b_nk(sB, 32, 0, ct * 16), acc[ct]);
    __syncthreads();
  }
#pragma unroll
  for (int ct = 0; ct < 4; ++ct) {
    int g = col0 + ct * 16 + (lane & 15);
    float bias = b1[g] + b2[g];
    int rb = row0 + wv * 16 + ((lane & 16) ? 8 : 0);
#pragma unroll
    for (int rr = 0; rr < 8; ++rr) C[(size_t)(rb + rr) * NG + g] = acc[ct][rr] + bias;
  }
}

// ---------------- persistent LSTM recurrence: 1 block, 8 waves ---------------
// per step: G(32,512) = xp[:,t,:] + H(32,128) @ Whh^T ; gates ; h,c update.
// Whh B-fragments live in VGPRs (16 v16h per wave) for all 128 steps.
__global__ void lstm_kernel(const float* __restrict__ xp,      // (B_,T_,NG)
                            const _Float16* __restrict__ whh,  // (NG,LH) f16
                            _Float16* __restrict__ hseq,       // (B_*T_,LH) f16
                            float* __restrict__ hfinal) {      // (B_,LH)
  __shared__ __align__(32) _Float16 sH[B_ * LH];  // [b][u] row-major
  __shared__ float sC[B_ * LH];
  __shared__ float sG[B_ * NG];
  int tid = threadIdx.x;  // 256
  int wv = tid >> 5;      // 0..7 : gate columns [wv*64, wv*64+64)
  int lane = lane_id();
  for (int i = tid; i < B_ * LH; i += 256) { sH[i] = (_Float16)0.f; sC[i] = 0.f; }
  v16h bf[4][4];
#pragma unroll
  for (int j = 0; j < 4; ++j)
#pragma unroll
    for (int k = 0; k < 4; ++k)
      bf[j][k] = pack_b_nk(whh, LH, k * 32, wv * 64 + j * 16);
  __syncthreads();
  for (int t = 0; t < T_; ++t) {
#pragma unroll
    for (int mt = 0; mt < 2; ++mt) {
      v16h a0 = pack_a_rm(sH, LH, mt * 16, 0);
      v16h a1 = pack_a_rm(sH, LH, mt * 16, 32);
      v16h a2 = pack_a_rm(sH, LH, mt * 16, 64);
      v16h a3 = pack_a_rm(sH, LH, mt * 16, 96);
#pragma unroll
      for (int j = 0; j < 4; ++j) {
        int g = wv * 64 + j * 16 + (lane & 15);
        int rb = mt * 16 + ((lane & 16) ? 8 : 0);
        v8f acc;
#pragma unroll
        for (int r = 0; r < 8; ++r)
          acc[r] = xp[((size_t)(rb + r) * T_ + t) * NG + g];
        acc = wmma16(a0, bf[j][0], acc);
        acc = wmma16(a1, bf[j][1], acc);
        acc = wmma16(a2, bf[j][2], acc);
        acc = wmma16(a3, bf[j][3], acc);
#pragma unroll
        for (int r = 0; r < 8; ++r) sG[(rb + r) * NG + g] = acc[r];
      }
    }
    __syncthreads();
    for (int i = tid; i < B_ * LH; i += 256) {
      int b = i >> 7, u = i & 127;
      float gi = sG[b * NG + u];
      float gf = sG[b * NG + LH + u];
      float gg = sG[b * NG + 2 * LH + u];
      float go = sG[b * NG + 3 * LH + u];
      gi = 1.f / (1.f + __expf(-gi));
      gf = 1.f / (1.f + __expf(-gf));
      gg = tanhf(gg);
      go = 1.f / (1.f + __expf(-go));
      float c = gf * sC[i] + gi * gg;
      float h = go * tanhf(c);
      sC[i] = c;
      sH[i] = (_Float16)h;
      hseq[((size_t)b * T_ + t) * LH + u] = (_Float16)h;
      if (t == T_ - 1) hfinal[i] = h;
    }
    __syncthreads();
  }
}

// ---------------- FC head: out = relu(h@fc1^T+b1) @ fc2^T + b2 ---------------
__global__ void head_kernel(const float* __restrict__ hf,
                            const float* __restrict__ w1, const float* __restrict__ b1,
                            const float* __restrict__ w2, const float* __restrict__ b2,
                            float* __restrict__ out) {
  __shared__ float sZ[B_ * 64];
  int tid = threadIdx.x;  // 256
  for (int i = tid; i < B_ * 64; i += 256) {
    int b = i >> 6, j = i & 63;
    float s = b1[j];
    for (int k = 0; k < LH; ++k) s += hf[b * LH + k] * w1[j * LH + k];
    sZ[i] = fmaxf(s, 0.f);
  }
  __syncthreads();
  for (int i = tid; i < B_ * NC; i += 256) {
    int b = i >> 4, c = i & 15;
    float s = b2[c];
    for (int k = 0; k < 64; ++k) s += sZ[b * 64 + k] * w2[c * 64 + k];
    out[i] = s;
  }
}

extern "C" void kernel_launch(void* const* d_in, const int* in_sizes, int n_in,
                              void* d_out, int out_size, void* d_ws, size_t ws_size,
                              hipStream_t stream) {
  (void)in_sizes; (void)n_in; (void)out_size; (void)ws_size;
  const float* x    = (const float*)d_in[0];
  const float* adj  = (const float*)d_in[1];
  const float* w1   = (const float*)d_in[2];
  const float* g1   = (const float*)d_in[3];
  const float* be1  = (const float*)d_in[4];
  const float* w2   = (const float*)d_in[5];
  const float* g2   = (const float*)d_in[6];
  const float* be2  = (const float*)d_in[7];
  const float* Wih0 = (const float*)d_in[8];
  const float* Whh0 = (const float*)d_in[9];
  const float* bih0 = (const float*)d_in[10];
  const float* bhh0 = (const float*)d_in[11];
  const float* Wih1 = (const float*)d_in[12];
  const float* Whh1 = (const float*)d_in[13];
  const float* bih1 = (const float*)d_in[14];
  const float* bhh1 = (const float*)d_in[15];
  const float* fc1w = (const float*)d_in[16];
  const float* fc1b = (const float*)d_in[17];
  const float* fc2w = (const float*)d_in[18];
  const float* fc2b = (const float*)d_in[19];
  float* out = (float*)d_out;

  char* base = (char*)d_ws;
  size_t off = 0;
  auto carve = [&](size_t bytes) -> void* {
    void* r = base + off;
    off = (off + bytes + 255) & ~(size_t)255;
    return r;
  };
  _Float16* adj16   = (_Float16*)carve((size_t)N_ * N_ * 2);
  float* cA1        = (float*)carve(64 * 4);
  float* cB1        = (float*)carve(64 * 4);
  float* cA2        = (float*)carve(64 * 4);
  float* cB2        = (float*)carve(64 * 4);
  float* buf0       = (float*)carve((size_t)CNT * 64 * 4);    // 64 MB
  float* buf1       = (float*)carve((size_t)CNT * 64 * 4);    // 64 MB
  _Float16* lin16   = (_Float16*)carve((size_t)M_ * DIN * 2); // 32 MB
  _Float16* wih0_16 = (_Float16*)carve((size_t)NG * DIN * 2);
  _Float16* whh0_16 = (_Float16*)carve((size_t)NG * LH * 2);
  _Float16* wih1_16 = (_Float16*)carve((size_t)NG * LH * 2);
  _Float16* whh1_16 = (_Float16*)carve((size_t)NG * LH * 2);
  float* xp0        = (float*)carve((size_t)M_ * NG * 4);     // 8 MB
  _Float16* hseq0   = (_Float16*)carve((size_t)M_ * LH * 2);
  float* xp1        = (float*)carve((size_t)M_ * NG * 4);
  _Float16* hseq1   = (_Float16*)carve((size_t)M_ * LH * 2);
  float* hfin0      = (float*)carve((size_t)B_ * LH * 4);
  float* hfin1      = (float*)carve((size_t)B_ * LH * 4);

  adjnorm_kernel<<<1, 64, 0, stream>>>(adj, adj16);
  int n0 = NG * DIN, n1 = NG * LH;
  f32_to_f16_kernel<<<(n0 + 255) / 256, 256, 0, stream>>>(Wih0, wih0_16, n0);
  f32_to_f16_kernel<<<(n1 + 255) / 256, 256, 0, stream>>>(Whh0, whh0_16, n1);
  f32_to_f16_kernel<<<(n1 + 255) / 256, 256, 0, stream>>>(Wih1, wih1_16, n1);
  f32_to_f16_kernel<<<(n1 + 255) / 256, 256, 0, stream>>>(Whh1, whh1_16, n1);

  // GCN layer 1
  gcn_kernel<<<M_, 128, 0, stream>>>(x, FIN, w1, adj16, buf0);
  bn_stats_kernel<<<64, 256, 0, stream>>>(buf0, g1, be1, cA1, cB1);
  bn_relu_kernel<<<CNT * 64 / 256, 256, 0, stream>>>(buf0, cA1, cB1, buf1);
  // GCN layer 2 + residual + pack to LSTM layout
  gcn_kernel<<<M_, 128, 0, stream>>>(buf1, GH, w2, adj16, buf0);
  bn_stats_kernel<<<64, 256, 0, stream>>>(buf0, g2, be2, cA2, cB2);
  bn_relu_add_pack_kernel<<<CNT * 64 / 256, 256, 0, stream>>>(buf0, cA2, cB2, buf1, lin16);

  dim3 gg(M_ / 64, NG / 64);
  gemm_xp_kernel<<<gg, 128, 0, stream>>>(lin16, wih0_16, bih0, bhh0, xp0, DIN);
  lstm_kernel<<<1, 256, 0, stream>>>(xp0, whh0_16, hseq0, hfin0);
  gemm_xp_kernel<<<gg, 128, 0, stream>>>(hseq0, wih1_16, bih1, bhh1, xp1, LH);
  lstm_kernel<<<1, 256, 0, stream>>>(xp1, whh1_16, hseq1, hfin1);
  head_kernel<<<1, 256, 0, stream>>>(hfin1, fc1w, fc1b, fc2w, fc2b, out);
}
